// SimilarityTreeLSTM_29154238005594
// MI455X (gfx1250) — matrix operand order repaired and checked by reading, not compiled, not run on hardware
//
#include <hip/hip_runtime.h>
#include <hip/hip_bf16.h>
#include <math.h>

typedef __attribute__((ext_vector_type(16))) __bf16 v16bf;
typedef __attribute__((ext_vector_type(8)))  float  v8f;

#define MEMD 256
#define INP  300
#define INPAD 320
#define LSEQ 256
#define MROWS 511

__device__ inline float sigf(float x) { return 1.0f / (1.0f + expf(-x)); }

// ---------------------------------------------------------------------------
// WMMA GEMM: C(MxN,f32) = A(MxK,f32) @ op(B). One wave computes a 16x64 slab
// (4 accumulators share one A fragment -> 4x fewer A loads/converts).
// B_KN=false: B is bf16 weights stored N x K row-major (C = A @ W^T).
// B_KN=true : B is stored K x N row-major (C = A @ B); B_F32 selects f32/bf16.
// K multiple of 32; N multiple of 64; M guarded per lane (EXEC full at WMMA).
// ---------------------------------------------------------------------------
template <bool B_F32, bool B_KN>
__global__ void gemm_wmma_kernel(const float* __restrict__ A, const void* __restrict__ Bv,
                                 float* __restrict__ C,
                                 int M, int N, int K, int lda, int ldb, int ldc) {
  int tilesN = N >> 6;                       // 64-wide wave slabs
  int Mpad   = (M + 15) & ~15;
  int tiles  = (Mpad >> 4) * tilesN;
  int tile   = blockIdx.x * (blockDim.x >> 5) + (threadIdx.x >> 5);
  if (tile >= tiles) return;
  int tm   = tile / tilesN;
  int tn   = tile - tm * tilesN;
  int lane = threadIdx.x & 31;
  int half = lane >> 4;   // 0: lanes 0-15, 1: lanes 16-31
  int l16  = lane & 15;
  int m    = tm * 16 + l16;   // A row held by this lane
  int n0   = tn * 64 + l16;   // first B/C column held by this lane

  const float*  Bf = (const float*)Bv;
  const __bf16* Bb = (const __bf16*)Bv;

  v8f acc[4];
#pragma unroll
  for (int t = 0; t < 4; ++t) acc[t] = (v8f){0.f, 0.f, 0.f, 0.f, 0.f, 0.f, 0.f, 0.f};

  for (int kb = 0; kb < K; kb += 32) {
    // A fragment (16-bit A 16x32): half=0 -> K kb+0..15, half=1 -> kb+16..31
    v16bf a;
    if (m < M) {
      const float* Ar = A + (size_t)m * lda + kb + half * 16;
#pragma unroll
      for (int j = 0; j < 16; ++j) a[j] = (__bf16)Ar[j];
    } else {
#pragma unroll
      for (int j = 0; j < 16; ++j) a[j] = (__bf16)0.f;
    }
#pragma unroll
    for (int t = 0; t < 4; ++t) {
      int n = n0 + t * 16;
      // B fragment (32x16): lanes 0-15 hold K kb+0..15 of column n, lanes 16-31 kb+16..31
      v16bf b;
#pragma unroll
      for (int j = 0; j < 16; ++j) {
        int kk = kb + half * 16 + j;
        if (B_KN) {
          b[j] = B_F32 ? (__bf16)Bf[(size_t)kk * ldb + n] : Bb[(size_t)kk * ldb + n];
        } else {
          b[j] = B_F32 ? (__bf16)Bf[(size_t)n * ldb + kk] : Bb[(size_t)n * ldb + kk];
        }
      }
      acc[t] = __builtin_amdgcn_wmma_f32_16x16x32_bf16(false, a, false, b, (short)0, acc[t],
                                                       false, false);
    }
  }

  // C layout: lane half selects M 0-7 / 8-15, VGPR r = row within half, column = l16
#pragma unroll
  for (int t = 0; t < 4; ++t) {
    int n = n0 + t * 16;
#pragma unroll
    for (int r = 0; r < 8; ++r) {
      int mm = tm * 16 + half * 8 + r;
      if (mm < M) C[(size_t)mm * ldc + n] = acc[t][r];
    }
  }
}

// ---------------------------------------------------------------------------
// Prep kernels
// ---------------------------------------------------------------------------
__global__ void convert_w_kernel(const float* __restrict__ src, int srcStride, int srcOff,
                                 int srcK, __bf16* __restrict__ dst, int dstK, int total) {
  int i = blockIdx.x * blockDim.x + threadIdx.x;
  if (i >= total) return;
  int r = i / dstK, k = i - r * dstK;
  dst[i] = (__bf16)((k < srcK) ? src[(size_t)r * srcStride + srcOff + k] : 0.f);
}

__global__ void gather_embed_kernel(const float* __restrict__ emb, const int* __restrict__ idx,
                                    float* __restrict__ dst, int total) {
  int i = blockIdx.x * blockDim.x + threadIdx.x;
  if (i >= total) return;
  int r = i / INPAD, k = i - r * INPAD;
  dst[i] = (k < INP) ? emb[(size_t)idx[r] * INP + k] : 0.f;
}

__global__ void zero_kernel(float* __restrict__ p, int n) {
  int i = blockIdx.x * blockDim.x + threadIdx.x;
  if (i < n) p[i] = 0.f;
}

// ---------------------------------------------------------------------------
// Tree-LSTM elementwise kernels (h/c stored row-contiguously at level offsets)
// ---------------------------------------------------------------------------
__global__ void gates_root_kernel(const float* __restrict__ iou, const float* __restrict__ bx,
                                  const float* __restrict__ bh, float* __restrict__ h,
                                  float* __restrict__ c, int total) {
  int i = blockIdx.x * blockDim.x + threadIdx.x;
  if (i >= total) return;
  int r = i >> 8, d = i & 255;
  const float* row = iou + (size_t)r * 768;
  float gi = sigf(row[d] + bx[d] + bh[d]);
  float go = sigf(row[256 + d] + bx[256 + d] + bh[256 + d]);
  float gu = tanhf(row[512 + d] + bx[512 + d] + bh[512 + d]);
  float cc = gi * gu;
  c[i] = cc;
  h[i] = go * tanhf(cc);
}

__global__ void gates_level_kernel(const float* __restrict__ iou, const float* __restrict__ bh,
                                   const float* __restrict__ fcsum, float* __restrict__ h,
                                   float* __restrict__ c, int total) {
  int i = blockIdx.x * blockDim.x + threadIdx.x;
  if (i >= total) return;
  int r = i >> 8, d = i & 255;
  const float* row = iou + (size_t)r * 768;
  float gi = sigf(row[d] + bh[d]);
  float go = sigf(row[256 + d] + bh[256 + d]);
  float gu = tanhf(row[512 + d] + bh[512 + d]);
  float cc = gi * gu + fcsum[i];
  c[i] = cc;
  h[i] = go * tanhf(cc);
}

// fcsum[j,d] = sig(fpre[2j,d]+b)*c[2j,d] + sig(fpre[2j+1,d]+b)*c[2j+1,d]
__global__ void fc_pairsum_kernel(const float* __restrict__ fpre, const float* __restrict__ bfh,
                                  const float* __restrict__ cprev, float* __restrict__ fcsum,
                                  int total) {
  int i = blockIdx.x * blockDim.x + threadIdx.x;
  if (i >= total) return;
  int r = i >> 8, d = i & 255;
  float f0 = sigf(fpre[(size_t)(2 * r) * 256 + d] + bfh[d]);
  float f1 = sigf(fpre[(size_t)(2 * r + 1) * 256 + d] + bfh[d]);
  fcsum[i] = f0 * cprev[(size_t)(2 * r) * 256 + d] + f1 * cprev[(size_t)(2 * r + 1) * 256 + d];
}

__global__ void pair_sum_kernel(const float* __restrict__ hprev, float* __restrict__ hsum, int total) {
  int i = blockIdx.x * blockDim.x + threadIdx.x;
  if (i >= total) return;
  int r = i >> 8, d = i & 255;
  hsum[i] = hprev[(size_t)(2 * r) * 256 + d] + hprev[(size_t)(2 * r + 1) * 256 + d];
}

// ---------------------------------------------------------------------------
// Attention kernels
// ---------------------------------------------------------------------------
// scores[r,m] = sum_d tanh(hp[r,d] + mp[m,d] + b[d]) * wa[d]; one wave per m, 8 waves/block
__global__ void scores_kernel(const float* __restrict__ hp, const float* __restrict__ mp,
                              const float* __restrict__ b, const float* __restrict__ wa,
                              float* __restrict__ scores) {
  int wave = threadIdx.x >> 5;
  int lane = threadIdx.x & 31;
  int m = blockIdx.x * 8 + wave;
  int r = blockIdx.y;
  if (m >= MROWS) return;
  const float* hr = hp + (size_t)r * 256;
  const float* mr = mp + (size_t)m * 256;
  float sum = 0.f;
  for (int d = lane; d < 256; d += 32) sum += tanhf(hr[d] + mr[d] + b[d]) * wa[d];
  for (int o = 16; o > 0; o >>= 1) sum += __shfl_down(sum, o, 32);
  if (lane == 0) scores[(size_t)r * MROWS + m] = sum;
}

// per-row softmax over 511, write (1-s) into a 512-wide padded buffer + ssum
__global__ void softmax_kernel(const float* __restrict__ scores, float* __restrict__ oms,
                               float* __restrict__ ssum) {
  int r = blockIdx.x, t = threadIdx.x;  // 256 threads
  __shared__ float red[256];
  float a = scores[(size_t)r * MROWS + t];
  float b = (t < 255) ? scores[(size_t)r * MROWS + 256 + t] : -1e30f;
  red[t] = fmaxf(a, b);
  __syncthreads();
  for (int s = 128; s > 0; s >>= 1) { if (t < s) red[t] = fmaxf(red[t], red[t + s]); __syncthreads(); }
  float mx = red[0];
  __syncthreads();
  float e1 = expf(a - mx);
  float e2 = (t < 255) ? expf(b - mx) : 0.f;
  red[t] = e1 + e2;
  __syncthreads();
  for (int s = 128; s > 0; s >>= 1) { if (t < s) red[t] += red[t + s]; __syncthreads(); }
  float Z = red[0];
  __syncthreads();
  float s1 = e1 / Z, s2 = e2 / Z;
  oms[(size_t)r * 512 + t] = 1.f - s1;
  if (t < 255) oms[(size_t)r * 512 + 256 + t] = 1.f - s2;
  if (t == 255) oms[(size_t)r * 512 + 511] = 0.f;
  red[t] = s1 + s2;
  __syncthreads();
  for (int s = 128; s > 0; s >>= 1) { if (t < s) red[t] += red[t + s]; __syncthreads(); }
  if (t == 0) ssum[r] = red[0];
}

// h[r,d] = attnout[r,d] + ssum[r] * h[r,d]
__global__ void attn_finish_kernel(const float* __restrict__ attnout, const float* __restrict__ ssum,
                                   float* __restrict__ h, int total) {
  int i = blockIdx.x * blockDim.x + threadIdx.x;
  if (i >= total) return;
  int r = i >> 8;
  h[i] = attnout[i] + ssum[r] * h[i];
}

// ---------------------------------------------------------------------------
// Head kernels
// ---------------------------------------------------------------------------
__global__ void final_v_kernel(const float* __restrict__ matl, const float* __restrict__ mats2,
                               const float* __restrict__ mats1, const float* __restrict__ matr,
                               float* __restrict__ v) {
  int d = threadIdx.x;  // 256
  const size_t off = (size_t)510 * 256;
  float lh = tanhf(matl[off + d] + mats2[off + d]);
  float rh = tanhf(mats1[off + d] + matr[off + d]);
  v[d] = lh * rh;
  v[256 + d] = fabsf(lh - rh);
}

__global__ void hid_sig_kernel(const float* __restrict__ raw, const float* __restrict__ bwh,
                               float* __restrict__ hid) {
  int d = threadIdx.x;  // 128
  hid[d] = sigf(raw[d] + bwh[d]);
}

__global__ void final_out_kernel(const float* __restrict__ hid, const float* __restrict__ Wwp,
                                 const float* __restrict__ bwp, float* __restrict__ out) {
  __shared__ float logits[5];
  int t = threadIdx.x;  // 32
  if (t < 5) {
    float s = bwp[t];
    for (int d = 0; d < 128; ++d) s += hid[d] * Wwp[t * 128 + d];
    logits[t] = s;
  }
  __syncthreads();
  if (t == 0) {
    float mx = logits[0];
    for (int c = 1; c < 5; ++c) mx = fmaxf(mx, logits[c]);
    float Z = 0.f;
    for (int c = 0; c < 5; ++c) Z += expf(logits[c] - mx);
    float lz = logf(Z) + mx;
    for (int c = 0; c < 5; ++c) out[c] = logits[c] - lz;
  }
}

// ---------------------------------------------------------------------------
// Host orchestration
// ---------------------------------------------------------------------------
static inline dim3 ew(int total) { return dim3((unsigned)((total + 255) / 256)); }

static void gemm_bf16(const float* A, const void* B, float* C, int M, int N, int K,
                      int lda, int ldb, int ldc, hipStream_t s) {
  int Mpad = (M + 15) & ~15;
  int tiles = (Mpad >> 4) * (N >> 6);
  gemm_wmma_kernel<false, false><<<dim3((unsigned)((tiles + 3) / 4)), dim3(128), 0, s>>>(
      A, B, C, M, N, K, lda, ldb, ldc);
}
static void gemm_f32kn(const float* A, const float* B, float* C, int M, int N, int K,
                       int lda, int ldb, int ldc, hipStream_t s) {
  int Mpad = (M + 15) & ~15;
  int tiles = (Mpad >> 4) * (N >> 6);
  gemm_wmma_kernel<true, true><<<dim3((unsigned)((tiles + 3) / 4)), dim3(128), 0, s>>>(
      A, (const void*)B, C, M, N, K, lda, ldb, ldc);
}

extern "C" void kernel_launch(void* const* d_in, const int* in_sizes, int n_in,
                              void* d_out, int out_size, void* d_ws, size_t ws_size,
                              hipStream_t stream) {
  (void)in_sizes; (void)n_in; (void)out_size; (void)ws_size;
  const int*   l_idx   = (const int*)d_in[0];
  const int*   r_idx   = (const int*)d_in[1];
  const float* emb     = (const float*)d_in[2];
  const float* W_ioux  = (const float*)d_in[3];
  const float* b_ioux  = (const float*)d_in[4];
  const float* W_iouh  = (const float*)d_in[5];
  const float* b_iouh  = (const float*)d_in[6];
  const float* W_fh    = (const float*)d_in[9];
  const float* b_fh    = (const float*)d_in[10];
  const float* Wa      = (const float*)d_in[11];
  const float* W_attnh = (const float*)d_in[12];
  const float* b_attnh = (const float*)d_in[13];
  const float* W_wh    = (const float*)d_in[14];
  const float* b_wh    = (const float*)d_in[15];
  const float* W_wp    = (const float*)d_in[16];
  const float* b_wp    = (const float*)d_in[17];
  float* out = (float*)d_out;

  // --- workspace bump allocator ---
  char* p = (char*)d_ws;
  auto alloc = [&](size_t bytes) -> void* {
    void* r = (void*)p;
    p += (bytes + 255) & ~(size_t)255;
    return r;
  };
  __bf16* Wioux_bf = (__bf16*)alloc((size_t)768 * INPAD * 2);
  __bf16* Wiouh_bf = (__bf16*)alloc((size_t)768 * 256 * 2);
  __bf16* Wfh_bf   = (__bf16*)alloc((size_t)256 * 256 * 2);
  __bf16* W1_bf    = (__bf16*)alloc((size_t)256 * 256 * 2);
  __bf16* W2_bf    = (__bf16*)alloc((size_t)256 * 256 * 2);
  __bf16* Wwh_bf   = (__bf16*)alloc((size_t)128 * 512 * 2);
  float* lx     = (float*)alloc((size_t)LSEQ * INPAD * 4);
  float* rx     = (float*)alloc((size_t)LSEQ * INPAD * 4);
  float* mat_l  = (float*)alloc((size_t)512 * 256 * 4);  // 511 rows + zero pad row
  float* mat_r  = (float*)alloc((size_t)512 * 256 * 4);
  float* mat_s1 = (float*)alloc((size_t)512 * 256 * 4);  // pass2 (rx attended)
  float* mat_s2 = (float*)alloc((size_t)512 * 256 * 4);  // pass4 (lx attended)
  float* cbuf   = (float*)alloc((size_t)512 * 256 * 4);
  float* mpbuf  = (float*)alloc((size_t)512 * 256 * 4);  // mat_proj
  float* ioubuf = (float*)alloc((size_t)256 * 768 * 4);
  float* hsum   = (float*)alloc((size_t)128 * 256 * 4);
  float* fbuf   = (float*)alloc((size_t)256 * 256 * 4);
  float* fcsum  = (float*)alloc((size_t)128 * 256 * 4);
  float* hpbuf  = (float*)alloc((size_t)256 * 256 * 4);
  float* scbuf  = (float*)alloc((size_t)256 * MROWS * 4);
  float* omsbuf = (float*)alloc((size_t)256 * 512 * 4);
  float* ssumb  = (float*)alloc((size_t)256 * 4);
  float* aobuf  = (float*)alloc((size_t)256 * 256 * 4);
  float* vbuf   = (float*)alloc((size_t)512 * 4);
  float* hidraw = (float*)alloc((size_t)128 * 4);
  float* hidbuf = (float*)alloc((size_t)128 * 4);

  // --- prep: bf16 weights, embedding gathers, zero pad rows ---
  convert_w_kernel<<<ew(768 * INPAD), 256, 0, stream>>>(W_ioux, INP, 0, INP, Wioux_bf, INPAD, 768 * INPAD);
  convert_w_kernel<<<ew(768 * 256), 256, 0, stream>>>(W_iouh, 256, 0, 256, Wiouh_bf, 256, 768 * 256);
  convert_w_kernel<<<ew(256 * 256), 256, 0, stream>>>(W_fh, 256, 0, 256, Wfh_bf, 256, 256 * 256);
  convert_w_kernel<<<ew(256 * 256), 256, 0, stream>>>(W_attnh, 512, 0, 256, W1_bf, 256, 256 * 256);
  convert_w_kernel<<<ew(256 * 256), 256, 0, stream>>>(W_attnh, 512, 256, 256, W2_bf, 256, 256 * 256);
  convert_w_kernel<<<ew(128 * 512), 256, 0, stream>>>(W_wh, 512, 0, 512, Wwh_bf, 512, 128 * 512);
  gather_embed_kernel<<<ew(LSEQ * INPAD), 256, 0, stream>>>(emb, l_idx, lx, LSEQ * INPAD);
  gather_embed_kernel<<<ew(LSEQ * INPAD), 256, 0, stream>>>(emb, r_idx, rx, LSEQ * INPAD);
  zero_kernel<<<1, 256, 0, stream>>>(mat_l + (size_t)511 * 256, 256);
  zero_kernel<<<1, 256, 0, stream>>>(mat_r + (size_t)511 * 256, 256);
  zero_kernel<<<1, 256, 0, stream>>>(mat_s1 + (size_t)511 * 256, 256);
  zero_kernel<<<1, 256, 0, stream>>>(mat_s2 + (size_t)511 * 256, 256);

  static const int offs[9] = {0, 256, 384, 448, 480, 496, 504, 508, 510};

  auto attend = [&](float* hrows, int k, const float* attnMat) {
    gemm_bf16(hrows, W1_bf, hpbuf, k, 256, 256, 256, 256, 256, stream);
    scores_kernel<<<dim3((MROWS + 7) / 8, (unsigned)k), 256, 0, stream>>>(hpbuf, mpbuf, b_attnh,
                                                                          Wa, scbuf);
    softmax_kernel<<<(unsigned)k, 256, 0, stream>>>(scbuf, omsbuf, ssumb);
    gemm_f32kn(omsbuf, attnMat, aobuf, k, 256, 512, 512, 256, 256, stream);
    attn_finish_kernel<<<ew(k * 256), 256, 0, stream>>>(aobuf, ssumb, hrows, k * 256);
  };

  auto tree_pass = [&](const float* X, float* matBuf, const float* attnMat) {
    if (attnMat)  // mat_proj = mat @ W2^T, once per pass
      gemm_bf16(attnMat, W2_bf, mpbuf, MROWS, 256, 256, 256, 256, 256, stream);
    // root: iou = X @ W_ioux^T (+ b_ioux + b_iouh in gates)
    gemm_bf16(X, Wioux_bf, ioubuf, LSEQ, 768, INPAD, INPAD, INPAD, 768, stream);
    gates_root_kernel<<<ew(LSEQ * 256), 256, 0, stream>>>(ioubuf, b_ioux, b_iouh, matBuf, cbuf, LSEQ * 256);
    if (attnMat) attend(matBuf, LSEQ, attnMat);
    for (int lvl = 1; lvl <= 8; ++lvl) {
      int n = LSEQ >> (lvl - 1);
      int k = LSEQ >> lvl;
      float* hprev = matBuf + (size_t)offs[lvl - 1] * 256;
      float* cprev = cbuf + (size_t)offs[lvl - 1] * 256;
      float* hcur = matBuf + (size_t)offs[lvl] * 256;
      float* ccur = cbuf + (size_t)offs[lvl] * 256;
      gemm_bf16(hprev, Wfh_bf, fbuf, n, 256, 256, 256, 256, 256, stream);       // f_pre = ch @ W_fh^T
      fc_pairsum_kernel<<<ew(k * 256), 256, 0, stream>>>(fbuf, b_fh, cprev, fcsum, k * 256);
      pair_sum_kernel<<<ew(k * 256), 256, 0, stream>>>(hprev, hsum, k * 256);
      gemm_bf16(hsum, Wiouh_bf, ioubuf, k, 768, 256, 256, 256, 768, stream);    // iou = hsum @ W_iouh^T
      gates_level_kernel<<<ew(k * 256), 256, 0, stream>>>(ioubuf, b_iouh, fcsum, hcur, ccur, k * 256);
      if (attnMat) attend(hcur, k, attnMat);
    }
  };

  tree_pass(lx, mat_l, nullptr);   // lh1, mat_l
  tree_pass(rx, mat_s1, mat_l);    // rh1 (attends over mat_l)
  tree_pass(rx, mat_r, nullptr);   // rh2, mat_r
  tree_pass(lx, mat_s2, mat_r);    // lh2 (attends over mat_r)

  final_v_kernel<<<1, 256, 0, stream>>>(mat_l, mat_s2, mat_s1, mat_r, vbuf);
  gemm_bf16(vbuf, Wwh_bf, hidraw, 1, 128, 512, 512, 512, 128, stream);
  hid_sig_kernel<<<1, 128, 0, stream>>>(hidraw, b_wh, hidbuf);
  final_out_kernel<<<1, 32, 0, stream>>>(hidbuf, W_wp, b_wp, out);
}
